// Experts_50646254355292
// MI455X (gfx1250) — compile-verified, matrix operand order checked by reading, not run on hardware
//
#include <hip/hip_runtime.h>
#include <hip/hip_fp16.h>

typedef _Float16 half_t;
typedef __attribute__((ext_vector_type(16))) _Float16 v16h;
typedef __attribute__((ext_vector_type(8)))  _Float16 v8h;
typedef __attribute__((ext_vector_type(4)))  _Float16 v4h;
typedef __attribute__((ext_vector_type(2)))  _Float16 v2h;
typedef __attribute__((ext_vector_type(8)))  float    v8f;

#define T_TOK 512
#define H_DIM 2880
#define I_DIM 2880
#define TWOI  5760
#define E_NUM 8
#define K_TOP 4

#define BM 128
#define BN 192   // divides both 5760 and 2880 -> no N tails anywhere
#define BK 32
#define AS 40    // padded k-stride (halfs); 80B rows keep 16B-aligned frag chunks

#define NB_CHUNK 3   // B: 3 chunks/thread, each 2 k-rows x 4 cols (32*192/8/256)

// ---------------------------------------------------------------------------
// Routing: combine weights, per-expert compacted token lists (deterministic,
// no atomics — serial scan by 8 expert-threads after a barrier).
// ---------------------------------------------------------------------------
__global__ void route_kernel(const int* __restrict__ ridx,
                             const float* __restrict__ rw,
                             float* __restrict__ combine,
                             int* __restrict__ pos,
                             int* __restrict__ list,
                             int* __restrict__ rows) {
  int t = threadIdx.x;  // 512 threads, one per token
  int i0 = ridx[t*K_TOP+0], i1 = ridx[t*K_TOP+1];
  int i2 = ridx[t*K_TOP+2], i3 = ridx[t*K_TOP+3];
  #pragma unroll
  for (int e = 0; e < E_NUM; ++e) {
    int c = (i0==e) + (i1==e) + (i2==e) + (i3==e);
    combine[t*E_NUM+e] = rw[t*E_NUM+e] * (float)c;
    pos[t*E_NUM+e] = -1;
  }
  __syncthreads();
  if (t < E_NUM) {
    int r = 0;
    for (int tt = 0; tt < T_TOK; ++tt) {
      int c = (ridx[tt*K_TOP+0]==t) + (ridx[tt*K_TOP+1]==t) +
              (ridx[tt*K_TOP+2]==t) + (ridx[tt*K_TOP+3]==t);
      if (c > 0) { pos[tt*E_NUM+t] = r; list[t*T_TOK + r] = tt; r++; }
    }
    rows[t] = r;
  }
}

// ---------------------------------------------------------------------------
// Per-expert tiled GEMM, f32->f16 conversion into LDS, v_wmma_f32_16x16x32_f16
// with f32 accumulate, double-buffered LDS, compile-time specialized A path.
// AMODE 0: A = gather(hidden, list[e]) f32 -> out f16 (raw gate_up, no bias)
// AMODE 1: A = hbuf[e] f16                 -> out f32 (+ down bias)
// No guards in the hot loop: rows >= Me gather a clamped (valid) row and
// produce garbage output rows that are never read downstream.
// Requires N % BN == 0 and Kdim % BK == 0.
// ---------------------------------------------------------------------------
template <int AMODE>
__global__ __launch_bounds__(256) void moe_gemm_kernel(
    const float* __restrict__ Bw,      // [E, Kdim, N] weights (fp32)
    const float* __restrict__ hidden,  // [T, Kdim] (AMODE 0)
    const half_t* __restrict__ Af16,   // [E, T, Kdim] (AMODE 1)
    const int* __restrict__ list,
    const int* __restrict__ rows,
    const float* __restrict__ bias,    // [E, N] (AMODE 1)
    float* __restrict__ outF32,        // [E, T, N] (AMODE 1)
    half_t* __restrict__ outF16,       // [E, T, N] (AMODE 0)
    int N, int Kdim) {
  __shared__ __align__(16) half_t Als[2][BM * AS];   // [m][k]
  __shared__ __align__(16) half_t Bls[2][BN * AS];   // [n][k] (transposed)

  const int e  = blockIdx.z;
  const int Me = rows[e];
  const int m0 = blockIdx.y * BM;
  if (m0 >= Me) return;                // uniform per block; implies Me >= 1
  const int n0  = blockIdx.x * BN;
  const int tid = threadIdx.x;
  const int nk  = Kdim / BK;           // 2880/32 = 90

  const float* Be = Bw + (size_t)e * Kdim * N;

  // ---- loop-invariant chunk descriptors (hoisted addressing) -------------
  // B: NB_CHUNK chunks/thread, each = 2 adjacent k-rows x 4 cols
  //    (packed v2h -> b32/2addr LDS stores). 48 col-groups per k-pair row.
  const float* bPtr[NB_CHUNK];         // global addr of k-row pair start
  int bBase[NB_CHUNK];                 // LDS offset: (ncg*4)*AS + 2*kp
  #pragma unroll
  for (int i = 0; i < NB_CHUNK; ++i) {
    int c   = tid + i * 256;           // 0..767
    int kp  = c / 48;                  // k-pair index 0..15
    int ncg = c - kp * 48;             // col group 0..47
    bPtr[i]  = Be + (size_t)(2 * kp) * N + (n0 + ncg * 4);
    bBase[i] = (ncg * 4) * AS + 2 * kp;
  }
  // A chunks: unconditional pointers (row clamped to a valid gather index).
  const float*  aPtr0[4];              // AMODE 0: gathered token row (+kc)
  const half_t* aPtr1[2];              // AMODE 1: f16 row (+kc)
  int aRow[4], aKc[4];
  if (AMODE == 0) {
    #pragma unroll
    for (int i = 0; i < 4; ++i) {
      int c   = tid + i * 256;         // 0..1023
      aRow[i] = c >> 3;                // 0..127
      aKc[i]  = (c & 7) * 4;           // 0..28
      int m = m0 + aRow[i];
      int mc = (m < Me) ? m : (Me - 1);        // clamp: avoid poison list entry
      aPtr0[i] = hidden + (size_t)list[e * T_TOK + mc] * Kdim + aKc[i];
    }
  } else {
    #pragma unroll
    for (int i = 0; i < 2; ++i) {
      int c   = tid + i * 256;         // 0..511
      aRow[i] = c >> 2;                // 0..127
      aKc[i]  = (c & 3) * 8;           // 0..24
      int m = m0 + aRow[i];            // always < T_TOK: in-bounds read
      aPtr1[i] = Af16 + ((size_t)e * T_TOK + m) * Kdim + aKc[i];
    }
  }

  float4 bReg[NB_CHUNK][2];
  float4 aReg0[4];
  v8h    aReg1[2];

  auto fetch = [&](int kt) {
    size_t kOff = (size_t)kt * BK;
    #pragma unroll
    for (int i = 0; i < NB_CHUNK; ++i) {
      const float* p0 = bPtr[i] + kOff * N;       // k-row 2*kp
      bReg[i][0] = *(const float4*)p0;
      bReg[i][1] = *(const float4*)(p0 + N);      // k-row 2*kp+1
    }
    if (AMODE == 0) {
      #pragma unroll
      for (int i = 0; i < 4; ++i)
        aReg0[i] = *(const float4*)(aPtr0[i] + kOff);
    } else {
      #pragma unroll
      for (int i = 0; i < 2; ++i)
        aReg1[i] = *(const v8h*)(aPtr1[i] + kOff);
    }
  };

  auto stage = [&](int s) {
    #pragma unroll
    for (int i = 0; i < NB_CHUNK; ++i) {   // B transpose-store: packed k-pairs
      const float* w0 = (const float*)&bReg[i][0];
      const float* w1 = (const float*)&bReg[i][1];
      #pragma unroll
      for (int j = 0; j < 4; ++j) {
        v2h p = { (half_t)w0[j], (half_t)w1[j] };
        *(v2h*)&Bls[s][bBase[i] + j * AS] = p;
      }
    }
    if (AMODE == 0) {
      #pragma unroll
      for (int i = 0; i < 4; ++i) {
        const float* vv = (const float*)&aReg0[i];
        v4h h;
        h[0] = (half_t)vv[0]; h[1] = (half_t)vv[1];
        h[2] = (half_t)vv[2]; h[3] = (half_t)vv[3];
        *(v4h*)&Als[s][aRow[i] * AS + aKc[i]] = h;
      }
    } else {
      #pragma unroll
      for (int i = 0; i < 2; ++i)
        *(v8h*)&Als[s][aRow[i] * AS + aKc[i]] = aReg1[i];
    }
  };

  // wave -> 32(M) x 96(N) sub-tile: 2 x 6 WMMA tiles (12 WMMA / 8 frag loads)
  const int lane = tid & 31;
  const int w    = tid >> 5;
  const int wm   = w >> 1;             // 0..3 -> 32-row strip
  const int wn   = w & 1;              // 0..1 -> 96-col strip
  const int lm   = lane & 15;
  const int g    = lane >> 4;          // lane half-group

  v8f acc[2][6] = {};

  union Fr { v16h v; v8h h[2]; };

  auto compute = [&](int s) {
    Fr af[2], bf[6];
    // ISA 16-bit A/B layout: lane half-group g holds K = 8g+j (j<8) and
    // K = 16+8g+(j-8) (j>=8): two contiguous 16B chunks per lane.
    #pragma unroll
    for (int a = 0; a < 2; ++a) {
      const half_t* base = &Als[s][(wm * 32 + a * 16 + lm) * AS + 8 * g];
      af[a].h[0] = *(const v8h*)base;
      af[a].h[1] = *(const v8h*)(base + 16);
    }
    #pragma unroll
    for (int b = 0; b < 6; ++b) {
      const half_t* base = &Bls[s][(wn * 96 + b * 16 + lm) * AS + 8 * g];
      bf[b].h[0] = *(const v8h*)base;
      bf[b].h[1] = *(const v8h*)(base + 16);
    }
    #pragma unroll
    for (int a = 0; a < 2; ++a)
      #pragma unroll
      for (int b = 0; b < 6; ++b)
        acc[a][b] = __builtin_amdgcn_wmma_f32_16x16x32_f16(
            false, af[a].v, false, bf[b].v, (short)0, acc[a][b], false, false);
  };

  // ---- software pipeline: nk-1 full iterations + tail compute ------------
  fetch(0);
  stage(0);
  __syncthreads();
  for (int kt = 0; kt < nk - 1; ++kt) {
    fetch(kt + 1);                     // overlap next-tile global loads
    compute(kt & 1);
    stage((kt + 1) & 1);
    __syncthreads();
  }
  compute((nk - 1) & 1);

  // Epilogue. C layout: col = lane&15, row = vgpr r + 8*(lane>>4).
  #pragma unroll
  for (int a = 0; a < 2; ++a) {
    #pragma unroll
    for (int b = 0; b < 6; ++b) {
      int col = n0 + wn * 96 + b * 16 + lm;
      float bval = (AMODE == 1) ? bias[(size_t)e * N + col] : 0.f;
      #pragma unroll
      for (int r = 0; r < 8; ++r) {
        int mrow = m0 + wm * 32 + a * 16 + g * 8 + r;
        float val = acc[a][b][r] + bval;
        if (AMODE == 0)
          outF16[((size_t)e * T_TOK + mrow) * N + col] = (half_t)val;
        else
          outF32[((size_t)e * T_TOK + mrow) * N + col] = val;
      }
    }
  }
}

// ---------------------------------------------------------------------------
// Activation: bias + clamp + GLU, f16 out.
// ---------------------------------------------------------------------------
__global__ __launch_bounds__(256) void act_kernel(
    const half_t* __restrict__ gub, const float* __restrict__ gu_bias,
    const int* __restrict__ rows, half_t* __restrict__ hbuf) {
  int e   = blockIdx.y;
  int idx = blockIdx.x * 256 + threadIdx.x;   // < T*I
  int row = idx / I_DIM;
  int i   = idx - row * I_DIM;
  if (row >= rows[e]) return;
  size_t base = ((size_t)e * T_TOK + row) * (size_t)TWOI;
  float gate = (float)gub[base + 2 * i]     + gu_bias[e * TWOI + 2 * i];
  float up   = (float)gub[base + 2 * i + 1] + gu_bias[e * TWOI + 2 * i + 1];
  gate = fminf(gate, 7.0f);
  up   = fminf(fmaxf(up, -7.0f), 7.0f);
  float glu = gate / (1.0f + __expf(-1.702f * gate));
  hbuf[((size_t)e * T_TOK + row) * I_DIM + i] = (half_t)((up + 1.0f) * glu);
}

// ---------------------------------------------------------------------------
// Final weighted combine (fixed expert order -> deterministic).
// ---------------------------------------------------------------------------
__global__ __launch_bounds__(256) void combine_kernel(
    const float* __restrict__ dout, const float* __restrict__ combine,
    const int* __restrict__ pos, float* __restrict__ y) {
  int idx = blockIdx.x * 256 + threadIdx.x;   // < T*H
  int t = idx / H_DIM;
  int h = idx - t * H_DIM;
  float acc = 0.f;
  #pragma unroll
  for (int e = 0; e < E_NUM; ++e) {
    int p = pos[t * E_NUM + e];
    if (p >= 0)
      acc += combine[t * E_NUM + e] * dout[((size_t)e * T_TOK + p) * H_DIM + h];
  }
  y[idx] = acc;
}

extern "C" void kernel_launch(void* const* d_in, const int* in_sizes, int n_in,
                              void* d_out, int out_size, void* d_ws, size_t ws_size,
                              hipStream_t stream) {
  (void)in_sizes; (void)n_in; (void)out_size; (void)ws_size;
  const float* hidden = (const float*)d_in[0];
  const int*   ridx   = (const int*)d_in[1];
  const float* rw     = (const float*)d_in[2];
  const float* gup    = (const float*)d_in[3];
  const float* gup_b  = (const float*)d_in[4];
  const float* dwn    = (const float*)d_in[5];
  const float* dwn_b  = (const float*)d_in[6];
  float* y = (float*)d_out;

  char* ws = (char*)d_ws;
  size_t off = 0;
  auto alloc = [&](size_t bytes) {
    size_t o = off; off += (bytes + 255) & ~(size_t)255; return o;
  };
  float*  combine = (float*)(ws + alloc((size_t)T_TOK * E_NUM * 4));
  int*    pos     = (int*)  (ws + alloc((size_t)T_TOK * E_NUM * 4));
  int*    list    = (int*)  (ws + alloc((size_t)E_NUM * T_TOK * 4));
  int*    rows    = (int*)  (ws + alloc((size_t)E_NUM * 4));
  half_t* gub     = (half_t*)(ws + alloc((size_t)E_NUM * T_TOK * TWOI * 2));
  half_t* hbuf    = (half_t*)(ws + alloc((size_t)E_NUM * T_TOK * I_DIM * 2));
  float*  dout    = (float*)(ws + alloc((size_t)E_NUM * T_TOK * H_DIM * 4));

  route_kernel<<<1, T_TOK, 0, stream>>>(ridx, rw, combine, pos, list, rows);

  // GEMM1: [Me,H] x [H,2I] -> raw gate_up (f16)
  moe_gemm_kernel<0><<<dim3(TWOI / BN, T_TOK / BM, E_NUM), 256, 0, stream>>>(
      gup, hidden, (const half_t*)nullptr, list, rows, (const float*)nullptr,
      (float*)nullptr, gub, TWOI, H_DIM);

  act_kernel<<<dim3((T_TOK * I_DIM) / 256, E_NUM), 256, 0, stream>>>(
      gub, gup_b, rows, hbuf);

  // GEMM2: [Me,I] x [I,H] + bias -> down output (f32)
  moe_gemm_kernel<1><<<dim3(H_DIM / BN, T_TOK / BM, E_NUM), 256, 0, stream>>>(
      dwn, (const float*)nullptr, hbuf, list, rows, dwn_b,
      dout, (half_t*)nullptr, H_DIM, I_DIM);

  combine_kernel<<<(T_TOK * H_DIM) / 256, 256, 0, stream>>>(dout, combine, pos, y);
}
